// MILAN_WoEdgeAug_27041114095831
// MI455X (gfx1250) — compile-verified, matrix operand order checked by reading, not run on hardware
//
#include <hip/hip_runtime.h>
#include <stdint.h>

#define HID   256
#define HEADS 8
#define HDIM  32
#define EPSN  1e-5f

typedef __attribute__((ext_vector_type(16))) __bf16 v16bf;
typedef __attribute__((ext_vector_type(8)))  float  v8f;

union U16 { v16bf v; uint32_t u[8]; };

__device__ __forceinline__ uint16_t f2bf(float f) {
  uint32_t u = __float_as_uint(f);
  u += 0x7FFFu + ((u >> 16) & 1u);          // round-to-nearest-even
  return (uint16_t)(u >> 16);
}
// monotone float <-> unsigned mapping for atomicMax-based segment max
__device__ __forceinline__ unsigned fenc(float f) {
  unsigned u = __float_as_uint(f);
  return (u & 0x80000000u) ? ~u : (u | 0x80000000u);
}
__device__ __forceinline__ float fdec(unsigned k) {
  unsigned u = (k & 0x80000000u) ? (k ^ 0x80000000u) : ~k;
  return __uint_as_float(u);
}

// Load one 16x32 bf16 tile in the WMMA 16-bit A layout from a row-major matrix.
// (B tiles use the same loader on pre-transposed weights: B layout == A layout of B^T.)
// Lane l (half = l>>4, m = l&15) holds row (row0+m); VGPR pairs cover
// K = k0 + half*8 + {0..7} and K = k0 + 16 + half*8 + {0..7} -> two b128 loads.
__device__ __forceinline__ v16bf load_tile16x32(const uint16_t* __restrict__ base,
                                                int ld, int row0, int k0, int lane) {
  const int half = lane >> 4;
  const int m    = lane & 15;
  const uint16_t* p = base + (size_t)(row0 + m) * ld + k0 + half * 8;
  uint4 lo = *(const uint4*)p;
  uint4 hi = *(const uint4*)(p + 16);
  U16 r;
  r.u[0] = lo.x; r.u[1] = lo.y; r.u[2] = lo.z; r.u[3] = lo.w;
  r.u[4] = hi.x; r.u[5] = hi.y; r.u[6] = hi.z; r.u[7] = hi.w;
  return r.v;
}

// ---------------- packing kernels ----------------
__global__ void pack_f32_to_bf16(const float* __restrict__ in, uint16_t* __restrict__ out, int n) {
  int i = blockIdx.x * blockDim.x + threadIdx.x;
  if (i < n) out[i] = f2bf(in[i]);
}

// wqkvt[col][k] = W[k][col&255] for col in [0,768): fused transposed QKV weight
__global__ void pack_wqkv_t(const float* __restrict__ WQ, const float* __restrict__ WK,
                            const float* __restrict__ WV, uint16_t* __restrict__ wt) {
  int i = blockIdx.x * blockDim.x + threadIdx.x;
  if (i >= 3 * HID * HID) return;
  int col = i >> 8, k = i & (HID - 1);
  const float* W = (col < HID) ? WQ : (col < 2 * HID) ? WK : WV;
  wt[i] = f2bf(W[(size_t)k * HID + (col & (HID - 1))]);
}

__global__ void pack_w_t(const float* __restrict__ W, uint16_t* __restrict__ wt) {
  int i = blockIdx.x * blockDim.x + threadIdx.x;
  if (i >= HID * HID) return;
  int col = i >> 8, k = i & (HID - 1);
  wt[i] = f2bf(W[(size_t)k * HID + col]);
}

// ---------------- WMMA GEMM: C[M,Ncols] = A[M,K](bf16) * Bt[Ncols,K]^T (+bias,+resid) ----------------
__global__ void gemm_wmma_bf16(const uint16_t* __restrict__ A, const uint16_t* __restrict__ Bt,
                               const float* __restrict__ bias, const float* __restrict__ resid,
                               float* __restrict__ C, int M, int Ncols, int K) {
  const int lane = threadIdx.x & 31;
  const int wave = threadIdx.x >> 5;
  const int mt   = blockIdx.x * (blockDim.x >> 5) + wave;   // wave-uniform guard: EXEC stays full
  const int nt   = blockIdx.y;
  if (mt * 16 >= M) return;

  v8f acc = {};
#pragma unroll
  for (int k0 = 0; k0 < 256; k0 += 32) {
    v16bf a = load_tile16x32(A,  K, mt * 16, k0, lane);
    v16bf b = load_tile16x32(Bt, K, nt * 16, k0, lane);
    acc = __builtin_amdgcn_wmma_f32_16x16x32_bf16(
        /*neg_a=*/false, a, /*neg_b=*/false, b,
        /*c_mod=*/(short)0, acc, /*reuse_a=*/false, /*reuse_b=*/false);
  }

  const int half = lane >> 4;
  const int col  = nt * 16 + (lane & 15);
  const float bv = bias ? bias[col] : 0.f;
#pragma unroll
  for (int r = 0; r < 8; ++r) {
    int row = mt * 16 + r + 8 * half;
    float v = acc[r] + bv;
    if (resid) v += resid[(size_t)row * Ncols + col];   // resid ld == Ncols when used
    C[(size_t)row * Ncols + col] = v;
  }
}

// ---------------- edge stage ----------------
// wave per edge; lane == head dim (HDIM==32==wave32)
__global__ void edge_scores(const float* __restrict__ qkv, const int* __restrict__ ei,
                            float* __restrict__ score, unsigned* __restrict__ segmax, int E) {
  const int lane = threadIdx.x & 31;
  const int e = blockIdx.x * (blockDim.x >> 5) + (threadIdx.x >> 5);
  if (e >= E) return;
  const int src = ei[e], dst = ei[E + e];
  const float* qd = qkv + (size_t)dst * (3 * HID);            // Q block
  const float* ks = qkv + (size_t)src * (3 * HID) + HID;      // K block
  const float sc = 0.17677669529663687f;                      // 1/sqrt(32)
#pragma unroll
  for (int h = 0; h < HEADS; ++h) {
    float v = qd[h * HDIM + lane] * ks[h * HDIM + lane];
#pragma unroll
    for (int off = 16; off; off >>= 1) v += __shfl_xor(v, off, 32);
    if (lane == 0) {
      float s = v * sc;
      score[(size_t)e * HEADS + h] = s;
      atomicMax(&segmax[(size_t)dst * HEADS + h], fenc(s));
    }
  }
}

__global__ void edge_expsum(float* __restrict__ sc_ex, const unsigned* __restrict__ segmax,
                            float* __restrict__ segsum, const int* __restrict__ ei, int E) {
  int i = blockIdx.x * blockDim.x + threadIdx.x;
  if (i >= E * HEADS) return;
  int e = i >> 3, h = i & 7;
  int dst = ei[E + e];
  float ex = expf(sc_ex[i] - fdec(segmax[(size_t)dst * HEADS + h]));
  sc_ex[i] = ex;
  atomicAdd(&segsum[(size_t)dst * HEADS + h], ex);
}

__global__ void edge_scatter(const float* __restrict__ qkv, const float* __restrict__ ex,
                             const float* __restrict__ segsum, const int* __restrict__ ei,
                             float* __restrict__ agg, int E) {
  const int lane = threadIdx.x & 31;
  const int e = blockIdx.x * (blockDim.x >> 5) + (threadIdx.x >> 5);
  if (e >= E) return;
  const int src = ei[e], dst = ei[E + e];
  const float* vs = qkv + (size_t)src * (3 * HID) + 2 * HID;  // V block
  float* ad = agg + (size_t)dst * HID;
#pragma unroll
  for (int h = 0; h < HEADS; ++h) {
    float alpha = ex[(size_t)e * HEADS + h] / segsum[(size_t)dst * HEADS + h];
    atomicAdd(&ad[h * HDIM + lane], alpha * vs[h * HDIM + lane]);
  }
}

// ---------------- GraphNorm + GELU ----------------
__global__ void col_stats(const float* __restrict__ hbuf, float* __restrict__ csum,
                          float* __restrict__ csumsq, int N) {
  int c = threadIdx.x;                       // 256 channels
  float s = 0.f, ss = 0.f;
  for (int r = blockIdx.x; r < N; r += gridDim.x) {
    float v = hbuf[(size_t)r * HID + c];
    s += v; ss += v * v;
  }
  atomicAdd(&csum[c], s);
  atomicAdd(&csumsq[c], ss);
}

__global__ void graphnorm_gelu(float* __restrict__ hbuf, const float* __restrict__ csum,
                               const float* __restrict__ csumsq, const float* __restrict__ gw,
                               const float* __restrict__ gb, const float* __restrict__ gs, int N) {
  int i = blockIdx.x * blockDim.x + threadIdx.x;
  if (i >= N * HID) return;
  int c = i & (HID - 1);
  float invN = 1.f / (float)N;
  float m = csum[c] * invN, msq = csumsq[c] * invN, s = gs[c];
  // E[(h - s*m)^2] = E[h^2] - (2s - s^2) * m^2
  float var = msq - (2.f * s - s * s) * m * m;
  float o = (hbuf[i] - s * m) * rsqrtf(var + EPSN) * gw[c] + gb[c];
  hbuf[i] = 0.5f * o * (1.f + erff(o * 0.70710678118654752f));
}

extern "C" void kernel_launch(void* const* d_in, const int* in_sizes, int n_in,
                              void* d_out, int out_size, void* d_ws, size_t ws_size,
                              hipStream_t stream) {
  const float* x  = (const float*)d_in[0];
  const float* WQ = (const float*)d_in[1];
  const float* WK = (const float*)d_in[2];
  const float* WV = (const float*)d_in[3];
  const float* WO = (const float*)d_in[4];
  const float* bo = (const float*)d_in[5];
  const float* gw = (const float*)d_in[6];
  const float* gb = (const float*)d_in[7];
  const float* gs = (const float*)d_in[8];
  const int*   ei = (const int*)d_in[9];
  const int N = in_sizes[0] / HID;          // 50000 (divisible by 16)
  const int E = in_sizes[9] / 2;            // 800000

  // workspace carving (256B aligned)
  char* p = (char*)d_ws;
  auto carve = [&](size_t bytes) { char* r = p; p += (bytes + 255) & ~(size_t)255; return r; };
  uint16_t* xb     = (uint16_t*)carve((size_t)N * HID * 2);
  uint16_t* wqkvt  = (uint16_t*)carve((size_t)3 * HID * HID * 2);
  uint16_t* woutt  = (uint16_t*)carve((size_t)HID * HID * 2);
  float*    qkv    = (float*)   carve((size_t)N * 3 * HID * 4);
  float*    ex     = (float*)   carve((size_t)E * HEADS * 4);
  unsigned* segmax = (unsigned*)carve((size_t)N * HEADS * 4);
  float*    segsum = (float*)   carve((size_t)N * HEADS * 4);
  float*    agg    = (float*)   carve((size_t)N * HID * 4);
  uint16_t* aggb   = (uint16_t*)carve((size_t)N * HID * 2);
  float*    csum   = (float*)   carve(HID * 4);
  float*    csumsq = (float*)   carve(HID * 4);
  float*    hbuf   = (float*)d_out;         // h lives in d_out, finalized in place

  hipMemsetAsync(segmax, 0, (size_t)N * HEADS * 4, stream);  // fenc order: 0u == -inf
  hipMemsetAsync(segsum, 0, (size_t)N * HEADS * 4, stream);
  hipMemsetAsync(agg,    0, (size_t)N * HID * 4,   stream);
  hipMemsetAsync(csum,   0, HID * 4, stream);
  hipMemsetAsync(csumsq, 0, HID * 4, stream);

  // pack
  pack_f32_to_bf16<<<(N * HID + 255) / 256, 256, 0, stream>>>(x, xb, N * HID);
  pack_wqkv_t<<<(3 * HID * HID + 255) / 256, 256, 0, stream>>>(WQ, WK, WV, wqkvt);
  pack_w_t<<<(HID * HID + 255) / 256, 256, 0, stream>>>(WO, woutt);

  // fused QKV GEMM: qkv[N,768] = x @ [WQ|WK|WV]
  const int Mtiles = N / 16;
  dim3 g1((Mtiles + 7) / 8, (3 * HID) / 16);
  gemm_wmma_bf16<<<g1, 256, 0, stream>>>(xb, wqkvt, nullptr, nullptr, qkv, N, 3 * HID, HID);

  // edge stage
  edge_scores<<<(E + 7) / 8, 256, 0, stream>>>(qkv, ei, ex, segmax, E);
  edge_expsum<<<(E * HEADS + 255) / 256, 256, 0, stream>>>(ex, segmax, segsum, ei, E);
  edge_scatter<<<(E + 7) / 8, 256, 0, stream>>>(qkv, ex, segsum, ei, agg, E);

  // output GEMM with fused bias + residual epilogue -> hbuf (= d_out)
  pack_f32_to_bf16<<<(N * HID + 255) / 256, 256, 0, stream>>>(agg, aggb, N * HID);
  dim3 g2((Mtiles + 7) / 8, HID / 16);
  gemm_wmma_bf16<<<g2, 256, 0, stream>>>(aggb, woutt, bo, x, hbuf, N, HID, HID);

  // GraphNorm (single-pass stats) + exact GELU, in place
  col_stats<<<256, HID, 0, stream>>>(hbuf, csum, csumsq, N);
  graphnorm_gelu<<<(N * HID + 255) / 256, 256, 0, stream>>>(hbuf, csum, csumsq, gw, gb, gs, N);
}
